// TransformerLayers_BidBird_24026047054599
// MI455X (gfx1250) — matrix operand order repaired and checked by reading, not compiled, not run on hardware
//
#include <hip/hip_runtime.h>

// ---------------- problem constants ----------------
#define LNUM   2
#define DMODEL 512
#define HN     8
#define DH     64
#define BSZ    16
#define RRAND  3
#define BB     2
#define SS     8192
#define NBLK   512
#define MMID   510
#define ROWS   (BB*SS)        // 16384
#define DFF    2048
#define CHUNK  2048           // FFN row chunk

typedef __bf16 bf16_t;
typedef __attribute__((ext_vector_type(16))) __bf16 v16bf;
typedef __attribute__((ext_vector_type(8)))  float  v8f;

static __device__ inline bf16_t f2bf(float f) {
    unsigned u = __builtin_bit_cast(unsigned, f);
    unsigned r = (u + 0x7FFFu + ((u >> 16) & 1u)) >> 16;   // round-to-nearest-even
    unsigned short us = (unsigned short)r;
    return __builtin_bit_cast(bf16_t, us);
}

static __device__ inline v8f vzero8() {
    v8f z;
#pragma unroll
    for (int i = 0; i < 8; ++i) z[i] = 0.f;
    return z;
}

static __device__ inline v8f wmma_bf16(v16bf a, v16bf b, v8f c) {
    // D(16x16,f32) = A(16x32,bf16) x B(32x16,bf16) + C
    return __builtin_amdgcn_wmma_f32_16x16x32_bf16(false, a, false, b, (short)0, c, false, false);
}

// Async DMA: 16 bytes / lane, global -> LDS, tracked on ASYNCcnt (no VGPR data path).
static __device__ inline void async_b128(unsigned ldsAddr, const float* gptr) {
    unsigned long long ga = (unsigned long long)(uintptr_t)gptr;
    asm volatile("global_load_async_to_lds_b128 %0, %1, off"
                 :: "v"(ldsAddr), "v"(ga) : "memory");
}

// ---- WMMA operand loaders ----
// A operand (16x32 bf16): lane 0-15 -> row=lane, K = {0..15}; lane 16-31 -> K = {16..31}
static __device__ inline v16bf load_a_bf(const bf16_t* __restrict__ src, int ld) {
    int lane = threadIdx.x & 31;
    int row = lane & 15;
    int cb  = (lane >> 4) * 16;
    v16bf a;
#pragma unroll
    for (int j = 0; j < 16; ++j) {
        int col = cb + j;
        a[j] = src[row * ld + col];
    }
    return a;
}
// Same layouts but reading fp32 LDS tiles (async copies land raw fp32), convert in-register.
static __device__ inline v16bf load_a_f32(const float* __restrict__ src, int ld) {
    int lane = threadIdx.x & 31;
    int row = lane & 15;
    int cb  = (lane >> 4) * 16;
    v16bf a;
#pragma unroll
    for (int j = 0; j < 16; ++j) {
        int col = cb + j;
        a[j] = (bf16_t)src[row * ld + col];
    }
    return a;
}
// B operand (32x16 bf16): lane col = lane&15, rows (lane>>4)*16 + j
static __device__ inline v16bf load_b_f32(const float* __restrict__ src, int ld) {
    int lane = threadIdx.x & 31;
    int col = lane & 15;
    int rb  = (lane >> 4) * 16;
    v16bf b;
#pragma unroll
    for (int j = 0; j < 16; ++j) b[j] = (bf16_t)src[(rb + j) * ld + col];
    return b;
}

// ---------------- generic WMMA GEMM, async double-buffered staging ----------------
// C[M,N] = act(A[M,K] @ W[K,N] + bias) (+resid). 64x64 tile, 4 waves, K-step 32.
// Tile k+1 streams global->LDS on ASYNCcnt while WMMAs consume tile k.
// OUT_MODE 0: fp32 row-major; OUT_MODE 1: bf16 into [b,h,s,d] attention layout.
template<int OUT_MODE, bool GELU_ACT, bool RESID>
__global__ __launch_bounds__(128) void gemm_wmma_k(
    const float* __restrict__ A, const float* __restrict__ W,
    const float* __restrict__ bias, const float* __restrict__ resid,
    void* __restrict__ outp, int N, int K)
{
    __shared__ float Asf[2][64][36];   // padded stride 36 (16B-chunk aligned)
    __shared__ float Bsf[2][32][68];   // padded stride 68
    const int rowBase = blockIdx.x * 64;
    const int colBase = blockIdx.y * 64;
    const int tid  = threadIdx.x;
    const int w    = tid >> 5;
    const int lane = tid & 31;

    // 8 async b128 per thread per tile: 4 chunks of A (64x32 f32), 4 chunks of B (32x64 f32)
    auto issue_tile = [&](int buf, int kt) {
#pragma unroll
        for (int i = 0; i < 4; ++i) {
            int c  = tid + i * 128;              // chunk id 0..511
            int ar = c >> 3, ac = (c & 7) * 4;   // A: row, col (4-float chunk)
            async_b128((unsigned)(uintptr_t)&Asf[buf][ar][ac],
                       &A[(size_t)(rowBase + ar) * K + kt + ac]);
            int br = c >> 4, bc = (c & 15) * 4;  // B: row, col
            async_b128((unsigned)(uintptr_t)&Bsf[buf][br][bc],
                       &W[(size_t)(kt + br) * N + colBase + bc]);
        }
    };

    v8f acc[4];
#pragma unroll
    for (int i = 0; i < 4; ++i) acc[i] = vzero8();

    issue_tile(0, 0);
    const int nIter = K / 32;
    for (int it = 0; it < nIter; ++it) {
        const int cur = it & 1;
        if (it + 1 < nIter) {
            issue_tile(1 - cur, (it + 1) * 32);
            // async loads complete in order: <=8 outstanding => previous tile landed
            asm volatile("s_wait_asynccnt 0x8" ::: "memory");
        } else {
            asm volatile("s_wait_asynccnt 0x0" ::: "memory");
        }
        __syncthreads();
        v16bf af = load_a_f32(&Asf[cur][w * 16][0], 36);
#pragma unroll
        for (int nt = 0; nt < 4; ++nt) {
            v16bf bfrag = load_b_f32(&Bsf[cur][0][nt * 16], 68);
            acc[nt] = wmma_bf16(af, bfrag, acc[nt]);
        }
        __syncthreads();   // tile fully consumed before next async overwrite
    }

    const int col16 = lane & 15, rh = lane >> 4;
#pragma unroll
    for (int nt = 0; nt < 4; ++nt) {
#pragma unroll
        for (int r = 0; r < 8; ++r) {
            int gr = rowBase + w * 16 + r + rh * 8;
            int gc = colBase + nt * 16 + col16;
            float val = acc[nt][r] + bias[gc];
            if (GELU_ACT) {
                float x = val;
                val = 0.5f * x * (1.f + tanhf(0.7978845608f * (x + 0.044715f * x * x * x)));
            }
            if (RESID) val += resid[(size_t)gr * N + gc];
            if (OUT_MODE == 0) {
                ((float*)outp)[(size_t)gr * N + gc] = val;
            } else {
                int b = gr >> 13, s = gr & (SS - 1);
                int h = gc >> 6,  dd = gc & 63;
                ((bf16_t*)outp)[((((size_t)b * HN + h) * SS + s) << 6) + dd] = f2bf(val);
            }
        }
    }
}

// ---------------- LayerNorm (wave per row of 512) ----------------
__global__ __launch_bounds__(256) void layernorm_k(
    const float* __restrict__ in, const float* __restrict__ g,
    const float* __restrict__ b, float* __restrict__ out)
{
    int row  = blockIdx.x * 8 + (threadIdx.x >> 5);
    int lane = threadIdx.x & 31;
    const float* rp = in + (size_t)row * DMODEL;
    float s = 0.f, sq = 0.f;
    for (int i = lane; i < DMODEL; i += 32) { float v = rp[i]; s += v; sq += v * v; }
#pragma unroll
    for (int off = 16; off; off >>= 1) { s += __shfl_xor(s, off, 32); sq += __shfl_xor(sq, off, 32); }
    float mu  = s * (1.f / DMODEL);
    float var = sq * (1.f / DMODEL) - mu * mu;
    float rs  = rsqrtf(var + 1e-5f);
    float* op = out + (size_t)row * DMODEL;
    for (int i = lane; i < DMODEL; i += 32) op[i] = (rp[i] - mu) * rs * g[i] + b[i];
}

// ---------------- flash attention per wave ----------------
// One 16-row Q block attends to nk keys (multiple of 32) selected by kmap.
// Scores and P@V through WMMA; online softmax on the 16x16 C layout.
template<class KMAP>
static __device__ void flash_attn_wave(
    const bf16_t* __restrict__ qh, const bf16_t* __restrict__ kh,
    const bf16_t* __restrict__ vh, const float* __restrict__ maskRow,
    int nk, int qrow, float* __restrict__ ctxB, int hIdx,
    bf16_t (*plds)[40], KMAP kmap)
{
    const int lane = threadIdx.x & 31;
    const int rowq = lane & 15;
    const int cb   = (lane >> 4) * 16;
    const int colk = lane & 15;
    const int rb   = (lane >> 4) * 16;
    const float scale = 0.125f;   // 1/sqrt(64)

    v16bf aq[2];
#pragma unroll
    for (int t = 0; t < 2; ++t) {
#pragma unroll
        for (int j = 0; j < 16; ++j) {
            int c = t * 32 + cb + j;
            aq[t][j] = qh[(size_t)(qrow + rowq) * DH + c];
        }
    }

    float Mr[8], Lr[8];
#pragma unroll
    for (int r = 0; r < 8; ++r) { Mr[r] = -1e30f; Lr[r] = 0.f; }
    v8f o[4];
#pragma unroll
    for (int i = 0; i < 4; ++i) o[i] = vzero8();

    for (int base = 0; base < nk; base += 32) {
        int key0 = kmap(base + colk);
        int key1 = kmap(base + 16 + colk);
        v8f s0 = vzero8(), s1 = vzero8();
#pragma unroll
        for (int t = 0; t < 2; ++t) {           // accumulate over d in 2 steps of 32
            v16bf b0, b1;
#pragma unroll
            for (int j = 0; j < 16; ++j) {
                int dd = t * 32 + rb + j;
                b0[j] = kh[(size_t)key0 * DH + dd];
                b1[j] = kh[(size_t)key1 * DH + dd];
            }
            s0 = wmma_bf16(aq[t], b0, s0);
            s1 = wmma_bf16(aq[t], b1, s1);
        }
        float bias0 = (maskRow[key0] - 1.f) * 1e9f;
        float bias1 = (maskRow[key1] - 1.f) * 1e9f;
#pragma unroll
        for (int r = 0; r < 8; ++r) {
            float x0 = s0[r] * scale + bias0;
            float x1 = s1[r] * scale + bias1;
            float mx = fmaxf(x0, x1);
#pragma unroll
            for (int off = 1; off < 16; off <<= 1) mx = fmaxf(mx, __shfl_xor(mx, off, 16));
            float mnew = fmaxf(Mr[r], mx);
            float ef = __expf(Mr[r] - mnew);
            float p0 = __expf(x0 - mnew);
            float p1 = __expf(x1 - mnew);
            float ps = p0 + p1;
#pragma unroll
            for (int off = 1; off < 16; off <<= 1) ps += __shfl_xor(ps, off, 16);
            Lr[r] = Lr[r] * ef + ps;
            Mr[r] = mnew;
#pragma unroll
            for (int nt = 0; nt < 4; ++nt) o[nt][r] *= ef;
            int prow = r + (lane >> 4) * 8;
            plds[prow][colk]      = f2bf(p0);
            plds[prow][16 + colk] = f2bf(p1);
        }
        asm volatile("s_wait_dscnt 0" ::: "memory");   // P stores visible before A-fragment reload
        v16bf pa = load_a_bf(&plds[0][0], 40);
#pragma unroll
        for (int nt = 0; nt < 4; ++nt) {
            v16bf bv;
#pragma unroll
            for (int j = 0; j < 16; ++j) {
                int key = kmap(base + rb + j);
                bv[j] = vh[(size_t)key * DH + nt * 16 + colk];
            }
            o[nt] = wmma_bf16(pa, bv, o[nt]);
        }
    }

#pragma unroll
    for (int nt = 0; nt < 4; ++nt) {
#pragma unroll
        for (int r = 0; r < 8; ++r) {
            float inv = 1.f / Lr[r];
            int srow = qrow + r + (lane >> 4) * 8;
            ctxB[(size_t)srow * DMODEL + hIdx * DH + nt * 16 + colk] = o[nt][r] * inv;
        }
    }
}

// global blocks (first + last) attend to all S keys
__global__ __launch_bounds__(32) void attn_global_k(
    const bf16_t* __restrict__ q, const bf16_t* __restrict__ k,
    const bf16_t* __restrict__ v, const float* __restrict__ maskPAD,
    float* __restrict__ ctx)
{
    __shared__ bf16_t plds[16][40];
    int wg = blockIdx.x;                 // b*16 + h*2 + g
    int g = wg & 1, h = (wg >> 1) & (HN - 1), b = wg >> 4;
    size_t headOff = ((size_t)b * HN + h) * SS * DH;
    int qrow = (g == 0) ? 0 : (NBLK - 1) * BSZ;
    flash_attn_wave(q + headOff, k + headOff, v + headOff, maskPAD + (size_t)b * SS,
                    SS, qrow, ctx + (size_t)b * SS * DMODEL, h, plds,
                    [](int i) { return i; });
}

// middle blocks: 48 window + 32 global + 48 random keys
__global__ __launch_bounds__(32) void attn_middle_k(
    const bf16_t* __restrict__ q, const bf16_t* __restrict__ k,
    const bf16_t* __restrict__ v, const float* __restrict__ maskPAD,
    const int* __restrict__ rand_idx, float* __restrict__ ctx)
{
    __shared__ bf16_t plds[16][40];
    int wg = blockIdx.x;
    int mi = wg % MMID;
    int h  = (wg / MMID) % HN;
    int b  = wg / (MMID * HN);
    int m  = mi + 1;
    int r0 = rand_idx[((size_t)h * MMID + mi) * RRAND + 0];
    int r1 = rand_idx[((size_t)h * MMID + mi) * RRAND + 1];
    int r2 = rand_idx[((size_t)h * MMID + mi) * RRAND + 2];
    size_t headOff = ((size_t)b * HN + h) * SS * DH;
    auto kmap = [=](int i) {
        if (i < 48) return (m - 1) * BSZ + i;               // window: blocks m-1,m,m+1
        if (i < 64) return i - 48;                          // global: block 0
        if (i < 80) return (NBLK - 1) * BSZ + (i - 64);     // global: last block
        int jj = i - 80; int rsel = jj >> 4;
        int blk = (rsel == 0) ? r0 : ((rsel == 1) ? r1 : r2);
        return blk * BSZ + (jj & 15);                       // random blocks
    };
    flash_attn_wave(q + headOff, k + headOff, v + headOff, maskPAD + (size_t)b * SS,
                    128, m * BSZ, ctx + (size_t)b * SS * DMODEL, h, plds, kmap);
}

// ---------------- driver ----------------
extern "C" void kernel_launch(void* const* d_in, const int* in_sizes, int n_in,
                              void* d_out, int out_size, void* d_ws, size_t ws_size,
                              hipStream_t stream)
{
    (void)in_sizes; (void)n_in; (void)out_size; (void)ws_size;
    const float* x_in  = (const float*)d_in[0];
    const float* maskP = (const float*)d_in[1];
    const int*   ridx  = (const int*)d_in[2];
    const float* Wq = (const float*)d_in[3];  const float* bq = (const float*)d_in[4];
    const float* Wk = (const float*)d_in[5];  const float* bk = (const float*)d_in[6];
    const float* Wv = (const float*)d_in[7];  const float* bv = (const float*)d_in[8];
    const float* Wo = (const float*)d_in[9];  const float* bo = (const float*)d_in[10];
    const float* g1 = (const float*)d_in[11]; const float* be1 = (const float*)d_in[12];
    const float* W1 = (const float*)d_in[13]; const float* b1 = (const float*)d_in[14];
    const float* W2 = (const float*)d_in[15]; const float* b2 = (const float*)d_in[16];
    const float* g2 = (const float*)d_in[17]; const float* be2 = (const float*)d_in[18];

    char* p = (char*)d_ws;
    bf16_t* Q  = (bf16_t*)p; p += (size_t)BB * HN * SS * DH * sizeof(bf16_t);
    bf16_t* Kb = (bf16_t*)p; p += (size_t)BB * HN * SS * DH * sizeof(bf16_t);
    bf16_t* Vb = (bf16_t*)p; p += (size_t)BB * HN * SS * DH * sizeof(bf16_t);
    float* CTX = (float*)p;  p += (size_t)ROWS * DMODEL * sizeof(float);
    float* Y   = (float*)p;  p += (size_t)ROWS * DMODEL * sizeof(float);
    float* XA  = (float*)p;  p += (size_t)ROWS * DMODEL * sizeof(float);
    float* X0  = (float*)p;  p += (size_t)ROWS * DMODEL * sizeof(float);
    float* HB  = (float*)p;  p += (size_t)CHUNK * DFF * sizeof(float);

    const dim3 gP(ROWS / 64, DMODEL / 64);        // projection-shaped GEMMs
    for (int i = 0; i < LNUM; ++i) {
        const float* xin = (i == 0) ? x_in : X0;
        size_t wOff = (size_t)i * DMODEL * DMODEL, bOff = (size_t)i * DMODEL;

        // Q,K,V projections -> bf16 [b,h,s,d]
        gemm_wmma_k<1, false, false><<<gP, 128, 0, stream>>>(xin, Wq + wOff, bq + bOff, nullptr, Q,  DMODEL, DMODEL);
        gemm_wmma_k<1, false, false><<<gP, 128, 0, stream>>>(xin, Wk + wOff, bk + bOff, nullptr, Kb, DMODEL, DMODEL);
        gemm_wmma_k<1, false, false><<<gP, 128, 0, stream>>>(xin, Wv + wOff, bv + bOff, nullptr, Vb, DMODEL, DMODEL);

        // BigBird attention -> CTX fp32 [b,s,D]
        attn_global_k<<<BB * HN * 2, 32, 0, stream>>>(Q, Kb, Vb, maskP, CTX);
        attn_middle_k<<<BB * HN * MMID, 32, 0, stream>>>(Q, Kb, Vb, maskP, ridx, CTX);

        // output projection + residual, then LN1
        gemm_wmma_k<0, false, true><<<gP, 128, 0, stream>>>(CTX, Wo + wOff, bo + bOff, xin, Y, DMODEL, DMODEL);
        layernorm_k<<<ROWS / 8, 256, 0, stream>>>(Y, g1 + bOff, be1 + bOff, XA);

        // FFN, chunked over rows (hidden buffer = CHUNK x DFF)
        for (int c = 0; c < ROWS / CHUNK; ++c) {
            const float* xa = XA + (size_t)c * CHUNK * DMODEL;
            gemm_wmma_k<0, true, false><<<dim3(CHUNK / 64, DFF / 64), 128, 0, stream>>>(
                xa, W1 + (size_t)i * DMODEL * DFF, b1 + (size_t)i * DFF, nullptr, HB, DFF, DMODEL);
            gemm_wmma_k<0, false, true><<<dim3(CHUNK / 64, DMODEL / 64), 128, 0, stream>>>(
                HB, W2 + (size_t)i * DFF * DMODEL, b2 + bOff, xa, Y + (size_t)c * CHUNK * DMODEL, DMODEL, DFF);
        }
        float* lnOut = (i == LNUM - 1) ? (float*)d_out : X0;
        layernorm_k<<<ROWS / 8, 256, 0, stream>>>(Y, g2 + bOff, be2 + bOff, lnOut);
    }
}
